// GRUModel_16054587753045
// MI455X (gfx1250) — matrix operand order repaired and checked by reading, not compiled, not run on hardware
//
#include <hip/hip_runtime.h>
#include <cstdint>
#include <cstddef>

#define T_DIM 4096
#define V_DIM 8192
#define H_DIM 1024

// ---- WMMA vector types (CDNA5, wave32) ----
typedef __attribute__((ext_vector_type(16))) __bf16 v16bf;
typedef __attribute__((ext_vector_type(8)))  __bf16 v8bf;
typedef __attribute__((ext_vector_type(8)))  float  v8f;
// ---- TDM descriptor vector types ----
typedef unsigned int u32x4 __attribute__((ext_vector_type(4)));
typedef int          i32x4 __attribute__((ext_vector_type(4)));
typedef int          i32x8 __attribute__((ext_vector_type(8)));

__device__ __forceinline__ unsigned short f2bf(float x) {
  unsigned u = __float_as_uint(x);
  if ((u & 0x7F800000u) == 0x7F800000u) return (unsigned short)(u >> 16); // inf/nan passthrough
  unsigned r = u + 0x7FFFu + ((u >> 16) & 1u);                            // round-to-nearest-even
  return (unsigned short)(r >> 16);
}

// Per-lane async global->LDS 16B copy (ASYNCcnt path, ISA 10 / 15.18.3 op 98).
__device__ __forceinline__ void async_copy_b128(const float* gsrc, float* ldst) {
  unsigned l = (unsigned)(size_t)ldst;                 // low 32 bits of generic LDS ptr = LDS addr
  unsigned long long g = (unsigned long long)(size_t)gsrc;
  asm volatile("global_load_async_to_lds_b128 %0, %1, off"
               :: "v"(l), "v"(g) : "memory");
}
__device__ __forceinline__ void wait_async0() {
#if __has_builtin(__builtin_amdgcn_s_wait_asynccnt)
  __builtin_amdgcn_s_wait_asynccnt(0);
#else
  asm volatile("s_wait_asynccnt 0x0" ::: "memory");
#endif
}

__global__ __launch_bounds__(256) void cvt_f32_to_bf16(const float* __restrict__ src,
                                                       unsigned short* __restrict__ dst,
                                                       size_t n) {
  for (size_t i = (size_t)blockIdx.x * blockDim.x + threadIdx.x; i < n;
       i += (size_t)gridDim.x * blockDim.x)
    dst[i] = f2bf(src[i]);
}

// C[M,N] = A[M,K] * B[N,K]^T (+ bias[N]); A,B bf16 row-major K-contiguous, C f32.
// 4x4 register-blocked: each wave computes a 64x64 block = 16 accumulators;
// per K-step (32) it loads 4 A-fragments + 4 B-fragments and issues 16
// v_wmma_f32_16x16x32_bf16 -> 4x fragment reuse vs 1-tile/wave.
// Per-lane fragment gathers follow ISA 7.12.2 (lane L, hi=L/16, lr=L%16):
//   A(16x32): a[j] = A[row=lr][(j<8?0:16) + 8*hi + (j&7)]  (two 16B chunks)
//   B(32x16): b[j] = Wrow[lr][(j<8?0:16) + 8*hi + (j&7)]  (two 16B chunks)
//   C/D     : c[p] = C[8*hi + p][lr]
__global__ __launch_bounds__(256) void gemm_nt_bf16_wmma(
    const unsigned short* __restrict__ Ah, const unsigned short* __restrict__ Bh,
    float* __restrict__ C, const float* __restrict__ bias,
    int M, int N, int K)
{
  const int wave  = blockIdx.x * (blockDim.x >> 5) + (threadIdx.x >> 5);
  const int ntn   = N >> 6;
  const int total = (M >> 6) * ntn;
  if (wave >= total) return;              // wave-uniform: EXEC stays all-ones for WMMA
  const int mt = (wave / ntn) << 6;
  const int nt = (wave % ntn) << 6;
  const int lane = threadIdx.x & 31;
  const int hi = lane >> 4;
  const int lr = lane & 15;
  const __bf16* Ab = (const __bf16*)Ah;
  const __bf16* Bb = (const __bf16*)Bh;
  const __bf16* Ar[4];
  const __bf16* Br[4];
  #pragma unroll
  for (int i = 0; i < 4; ++i) {
    Ar[i] = Ab + (size_t)(mt + 16 * i + lr) * K;
    Br[i] = Bb + (size_t)(nt + 16 * i + lr) * K;
  }
  v8f acc[4][4] = {};
  for (int k = 0; k < K; k += 32) {
    v16bf av[4], bv[4];
    #pragma unroll
    for (int i = 0; i < 4; ++i) {
      v8bf a0 = *(const v8bf*)(Ar[i] + k + 8 * hi);
      v8bf a1 = *(const v8bf*)(Ar[i] + k + 16 + 8 * hi);
      av[i] = __builtin_shufflevector(a0, a1, 0,1,2,3,4,5,6,7,8,9,10,11,12,13,14,15);
      v8bf b0 = *(const v8bf*)(Br[i] + k + 8 * hi);
      v8bf b1 = *(const v8bf*)(Br[i] + k + 16 + 8 * hi);
      bv[i] = __builtin_shufflevector(b0, b1, 0,1,2,3,4,5,6,7,8,9,10,11,12,13,14,15);
    }
    #pragma unroll
    for (int im = 0; im < 4; ++im)
      #pragma unroll
      for (int in = 0; in < 4; ++in)
        acc[im][in] = __builtin_amdgcn_wmma_f32_16x16x32_bf16(
            false, av[im], false, bv[in], (short)0, acc[im][in], false, false);
  }
  #pragma unroll
  for (int im = 0; im < 4; ++im) {
    #pragma unroll
    for (int in = 0; in < 4; ++in) {
      const float badd = bias ? bias[nt + 16 * in + lr] : 0.0f;
      float* Cp = C + (size_t)(mt + 16 * im + 8 * hi) * N + (nt + 16 * in + lr);
      #pragma unroll
      for (int p = 0; p < 8; ++p) Cp[(size_t)p * N] = acc[im][in][p] + badd;
    }
  }
}

// ---- persistent sequential GRU scan ----
#define NB 96            // 32 blocks each for Whz / Whr / Whk rows
#define RB_THREADS 256
#define ROWS 32          // rows of the HxH matrix owned per block (LDS-resident)

__device__ __forceinline__ void gbar(unsigned* ctr, unsigned target) {
  __threadfence();                         // make this block's global stores visible
  __syncthreads();
  if (threadIdx.x == 0) {
    atomicAdd(ctr, 1u);
    volatile unsigned* vc = (volatile unsigned*)ctr;
    while (*vc < target) __builtin_amdgcn_s_sleep(2);
  }
  __syncthreads();
  __threadfence();
}

__global__ __launch_bounds__(RB_THREADS) void gru_scan(
    const float* __restrict__ Whz, const float* __restrict__ Whr,
    const float* __restrict__ Whk,
    const float* __restrict__ xz, const float* __restrict__ xr,
    const float* __restrict__ xk,
    const float* __restrict__ h_init,
    float* __restrict__ hbuf, float* __restrict__ zbuf, float* __restrict__ rhbuf,
    unsigned short* __restrict__ hs_b, unsigned* __restrict__ ctr)
{
  extern __shared__ float smem[];
  float* sw = smem;                    // ROWS x H_DIM weight slice (dense; TDM-written)
  float* sh = smem + ROWS * H_DIM;     // H-vector staging (h or r*h)
  const int tid  = threadIdx.x;
  const int role = blockIdx.x >> 5;    // 0: z-rows, 1: r-rows, 2: k-rows
  const int bi   = blockIdx.x & 31;

  const float* wsrc =
      (role == 0 ? Whz : (role == 1 ? Whr : Whk)) + (size_t)bi * ROWS * H_DIM;

  // ---- one-shot weight-slice preload: Tensor Data Mover if available ----
#if __has_builtin(__builtin_amdgcn_tensor_load_to_lds)
  if (tid < 32) {                      // wave 0 issues the DMA (EXEC ignored by TDM)
    unsigned lds_base = (unsigned)(size_t)sw;
    unsigned long long ga = (unsigned long long)(size_t)wsrc;
    // D# group0: count=1, lds_addr, global_addr[56:0], type=2 ("image")
    u32x4 g0 = { 1u, lds_base,
                 (unsigned)(ga & 0xffffffffu),
                 (unsigned)((ga >> 32) & 0x01ffffffu) | (2u << 30) };
    // D# group1: data_size=4B; tensor_dim0=1024, tensor_dim1=1024;
    //            tile_dim0=1024, tile_dim1=32; tensor_dim0_stride=1024
    i32x8 g1 = { (int)(2u << 16),
                 (int)(1024u << 16),   // tensor_dim0[15:0] @ bits 63:48
                 (int)(1024u << 16),   // tensor_dim0 hi=0 | tensor_dim1[15:0]
                 (int)(1024u << 16),   // tensor_dim1 hi=0 | tile_dim0
                 32,                   // tile_dim1=32, tile_dim2=0
                 1024,                 // tensor_dim0_stride[31:0]
                 0, 0 };
    i32x4 gz = { 0, 0, 0, 0 };
  #if __clang_major__ >= 23
    i32x8 gz8 = { 0, 0, 0, 0, 0, 0, 0, 0 };
    __builtin_amdgcn_tensor_load_to_lds(g0, g1, gz, gz, gz8, 0);
  #else
    __builtin_amdgcn_tensor_load_to_lds(g0, g1, gz, gz, 0);
  #endif
    __builtin_amdgcn_s_wait_tensorcnt(0);
  }
#else
  for (int j = tid * 4; j < ROWS * H_DIM; j += RB_THREADS * 4)
    async_copy_b128(wsrc + j, sw + j);
  wait_async0();
#endif

  if (blockIdx.x == 0)
    for (int j = tid; j < H_DIM; j += RB_THREADS) hbuf[j] = h_init[j];

  unsigned tgt = NB;
  gbar(ctr, tgt);

  const int rl  = tid >> 3;            // local row 0..31 (8 lanes per row, same wave)
  const int oct = tid & 7;
  const int i   = bi * ROWS + rl;      // global row index 0..1023

  for (int t = 0; t < T_DIM; ++t) {
    const float* hc = hbuf + (size_t)(t & 1) * H_DIM;
    float*       hn = hbuf + (size_t)((t + 1) & 1) * H_DIM;

    if (role < 2) {                    // phase 1: u=Whz*h -> z ; v=Whr*h -> r*h
      async_copy_b128(hc + tid * 4, sh + tid * 4);   // stage h into LDS (async path)
      wait_async0();
      __syncthreads();
      float p = 0.f;
      const float* wr = sw + rl * H_DIM;
      #pragma unroll 8
      for (int j = 0; j < H_DIM / 8; ++j) {
        int c = oct + (j << 3);
        p = fmaf(wr[c], sh[c], p);
      }
      p += __shfl_down(p, 4, 8);
      p += __shfl_down(p, 2, 8);
      p += __shfl_down(p, 1, 8);
      if (oct == 0) {
        if (role == 0) {
          float z = 1.0f / (1.0f + __expf(-(xz[(size_t)t * H_DIM + i] + p)));
          zbuf[i] = z;
          hs_b[(size_t)t * H_DIM + i] = f2bf(sh[i]);   // output collects pre-update h_t
        } else {
          float r = 1.0f / (1.0f + __expf(-(xr[(size_t)t * H_DIM + i] + p)));
          rhbuf[i] = r * sh[i];
        }
      }
    }
    tgt += NB; gbar(ctr, tgt);

    if (role == 2) {                   // phase 2: w=Whk*(r*h) -> k -> h_new
      async_copy_b128(rhbuf + tid * 4, sh + tid * 4); // stage r*h into LDS (async path)
      wait_async0();
      __syncthreads();
      float p = 0.f;
      const float* wr = sw + rl * H_DIM;
      #pragma unroll 8
      for (int j = 0; j < H_DIM / 8; ++j) {
        int c = oct + (j << 3);
        p = fmaf(wr[c], sh[c], p);
      }
      p += __shfl_down(p, 4, 8);
      p += __shfl_down(p, 2, 8);
      p += __shfl_down(p, 1, 8);
      if (oct == 0) {
        float kk = tanhf(xk[(size_t)t * H_DIM + i] + p);
        float z  = zbuf[i];
        float hv = hc[i];
        hn[i] = (1.0f - z) * hv + z * kk;
      }
    }
    tgt += NB; gbar(ctr, tgt);
  }
}

extern "C" void kernel_launch(void* const* d_in, const int* in_sizes, int n_in,
                              void* d_out, int out_size, void* d_ws, size_t ws_size,
                              hipStream_t stream) {
  (void)in_sizes; (void)n_in; (void)out_size; (void)ws_size;
  const float* input = (const float*)d_in[0];
  const float* W_xz  = (const float*)d_in[1];
  const float* W_xr  = (const float*)d_in[2];
  const float* W_xk  = (const float*)d_in[3];
  const float* W_hz  = (const float*)d_in[4];
  const float* W_hr  = (const float*)d_in[5];
  const float* W_hk  = (const float*)d_in[6];
  const float* W_hy  = (const float*)d_in[7];
  const float* B_z   = (const float*)d_in[8];
  const float* B_r   = (const float*)d_in[9];
  const float* B_k   = (const float*)d_in[10];
  const float* h0    = (const float*)d_in[11];
  float* out = (float*)d_out;

  char* p = (char*)d_ws;
  auto take = [&](size_t bytes) {
    char* q = p; p += (bytes + 255) & ~(size_t)255; return q;
  };
  unsigned* ctr        = (unsigned*)take(sizeof(unsigned));
  float* hbuf          = (float*)take((size_t)2 * H_DIM * sizeof(float));
  float* zbuf          = (float*)take((size_t)H_DIM * sizeof(float));
  float* rhbuf         = (float*)take((size_t)H_DIM * sizeof(float));
  float* xproj         = (float*)take((size_t)3 * T_DIM * H_DIM * sizeof(float));
  unsigned short* hsb  = (unsigned short*)take((size_t)T_DIM * H_DIM * 2);
  unsigned short* inb  = (unsigned short*)take((size_t)T_DIM * V_DIM * 2);
  unsigned short* wxzb = (unsigned short*)take((size_t)H_DIM * V_DIM * 2);
  unsigned short* wxrb = (unsigned short*)take((size_t)H_DIM * V_DIM * 2);
  unsigned short* wxkb = (unsigned short*)take((size_t)H_DIM * V_DIM * 2);
  unsigned short* whyb = (unsigned short*)take((size_t)V_DIM * H_DIM * 2);

  hipMemsetAsync(ctr, 0, sizeof(unsigned), stream);   // barrier counter reset each launch

  cvt_f32_to_bf16<<<4096, 256, 0, stream>>>(input, inb, (size_t)T_DIM * V_DIM);
  cvt_f32_to_bf16<<<2048, 256, 0, stream>>>(W_xz, wxzb, (size_t)H_DIM * V_DIM);
  cvt_f32_to_bf16<<<2048, 256, 0, stream>>>(W_xr, wxrb, (size_t)H_DIM * V_DIM);
  cvt_f32_to_bf16<<<2048, 256, 0, stream>>>(W_xk, wxkb, (size_t)H_DIM * V_DIM);
  cvt_f32_to_bf16<<<2048, 256, 0, stream>>>(W_hy, whyb, (size_t)V_DIM * H_DIM);

  // Stage A: xz/xr/xk = input @ W^T + b   ([4096,8192]x[8192,1024], bf16 WMMA, 64x64/wave)
  const int wavesA  = (T_DIM / 64) * (H_DIM / 64);
  const int blocksA = wavesA / 8;
  gemm_nt_bf16_wmma<<<blocksA, 256, 0, stream>>>(inb, wxzb, xproj, B_z, T_DIM, H_DIM, V_DIM);
  gemm_nt_bf16_wmma<<<blocksA, 256, 0, stream>>>(inb, wxrb, xproj + (size_t)T_DIM * H_DIM,
                                                 B_r, T_DIM, H_DIM, V_DIM);
  gemm_nt_bf16_wmma<<<blocksA, 256, 0, stream>>>(inb, wxkb, xproj + (size_t)2 * T_DIM * H_DIM,
                                                 B_k, T_DIM, H_DIM, V_DIM);

  // Stage B: persistent sequential scan, LDS-resident recurrent weights (TDM/async preload)
  const size_t smemBytes = (size_t)(ROWS * H_DIM + H_DIM) * sizeof(float);  // 132 KB/WGP
  gru_scan<<<NB, RB_THREADS, smemBytes, stream>>>(
      W_hz, W_hr, W_hk,
      xproj, xproj + (size_t)T_DIM * H_DIM, xproj + (size_t)2 * T_DIM * H_DIM,
      h0, hbuf, zbuf, rhbuf, hsb, ctr);

  // Stage C: out = hs @ W_hy^T   ([4096,1024]x[1024,8192], bf16 WMMA, 64x64/wave)
  const int wavesC  = (T_DIM / 64) * (V_DIM / 64);
  const int blocksC = wavesC / 8;
  gemm_nt_bf16_wmma<<<blocksC, 256, 0, stream>>>(hsb, whyb, out, nullptr, T_DIM, V_DIM, H_DIM);
}
